// CausalSelfAttention_62294205661212
// MI455X (gfx1250) — matrix operand order repaired and checked by reading, not compiled, not run on hardware
//
#include <hip/hip_runtime.h>

#define DI __device__ __forceinline__

typedef __attribute__((ext_vector_type(16))) __bf16 bf16x16;
typedef __attribute__((ext_vector_type(8)))  __bf16 bf16x8;
typedef __attribute__((ext_vector_type(4)))  __bf16 bf16x4;
typedef __attribute__((ext_vector_type(8)))  float  f32x8;

constexpr int B_ = 4, T_ = 2048, C_ = 1024, H_ = 16, D_ = 64;
constexpr int M_ = B_ * T_;     // 8192 rows
constexpr int N1 = 3 * C_;      // 3072 qkv cols
constexpr float SCALE = 0.125f; // 1/sqrt(64)

// ---------- helpers ----------

DI __bf16 f2bf(float f) { return (__bf16)f; }  // native v_cvt on gfx1250

// A-matrix fragment, 16x32 bf16 (ISA 7.12.2): lane row = lane%16,
// khalf = lane/16: elems 0..7 at k = khalf*8 + 0..7, elems 8..15 at +16.
// p must point at (row base + khalf*8); 16B aligned.
DI bf16x16 ld_afrag(const __bf16* p) {
  bf16x8 lo = *(const bf16x8*)(p);
  bf16x8 hi = *(const bf16x8*)(p + 16);
  bf16x16 f;
#pragma unroll
  for (int i = 0; i < 8; ++i) { f[i] = lo[i]; f[i + 8] = hi[i]; }
  return f;
}

// B-matrix fragment, 32x16 bf16: lane col = lane%16, khalf = lane/16:
// elems 0..7 at k = khalf*8 + 0..7, elems 8..15 at +16. p points at
// (Bt row base + khalf*8) where Bt is the 16-col tile stored as [n][k].
DI bf16x16 ld_bfrag(const __bf16* p) {
  bf16x8 lo = *(const bf16x8*)(p);
  bf16x8 hi = *(const bf16x8*)(p + 16);
  bf16x16 f;
#pragma unroll
  for (int i = 0; i < 8; ++i) { f[i] = lo[i]; f[i + 8] = hi[i]; }
  return f;
}

DI f32x8 wmma_bf16(bf16x16 a, bf16x16 b, f32x8 c) {
  return __builtin_amdgcn_wmma_f32_16x16x32_bf16(
      /*neg_a=*/false, a, /*neg_b=*/false, b,
      /*c_mod=*/(short)0, c, /*reuse_a=*/false, /*reuse_b=*/false);
}

DI void cvt_store4(__bf16* dst, float4 v) {  // one 8B LDS store
  bf16x4 p = { f2bf(v.x), f2bf(v.y), f2bf(v.z), f2bf(v.w) };
  *(bf16x4*)dst = p;
}

// ---------- kernel 1: qkv = x @ w_qkv + b, scatter to [B,H,T,D] bf16 ----------

__global__ __launch_bounds__(256)
void qkv_gemm_kernel(const float* __restrict__ x, const float* __restrict__ w,
                     const float* __restrict__ bias,
                     __bf16* __restrict__ qb, __bf16* __restrict__ kb,
                     __bf16* __restrict__ vb) {
  __shared__ __bf16 At[128 * 64];   // [m][k]
  __shared__ __bf16 Bt[128 * 64];   // [n][k] (transposed stage)
  const int tid = threadIdx.x, lane = tid & 31, wv = tid >> 5;
  const int ln = lane & 15, hi = lane >> 4;
  const int n0 = blockIdx.x * 128, m0 = blockIdx.y * 128;
  const int wvM = (wv >> 2) * 64, wvN = (wv & 3) * 32;

  const int arow = tid >> 1, akoff = (tid & 1) * 32;   // 128 rows x 64 k
  const int bk = tid >> 2,  bnoff = (tid & 3) * 32;    // 64 k x 128 n

  f32x8 acc[4][2] = {};

  for (int kb0 = 0; kb0 < C_; kb0 += 64) {
    __syncthreads();
    { // stage A (fp32 -> bf16), packed 8B stores
      const float4* src = (const float4*)(x + (size_t)(m0 + arow) * C_ + kb0 + akoff);
      __bf16* dst = &At[arow * 64 + akoff];
#pragma unroll
      for (int q = 0; q < 8; ++q) cvt_store4(dst + q * 4, src[q]);
    }
    { // stage B transposed
      const float4* src = (const float4*)(w + (size_t)(kb0 + bk) * N1 + n0 + bnoff);
#pragma unroll
      for (int q = 0; q < 8; ++q) {
        float4 v = src[q];
        Bt[(bnoff + q * 4 + 0) * 64 + bk] = f2bf(v.x);
        Bt[(bnoff + q * 4 + 1) * 64 + bk] = f2bf(v.y);
        Bt[(bnoff + q * 4 + 2) * 64 + bk] = f2bf(v.z);
        Bt[(bnoff + q * 4 + 3) * 64 + bk] = f2bf(v.w);
      }
    }
    if (kb0 + 64 < C_) { // prefetch next K-step (global_prefetch_b8)
      __builtin_prefetch(x + (size_t)(m0 + arow) * C_ + kb0 + 64 + akoff, 0, 1);
      __builtin_prefetch(w + (size_t)(kb0 + 64 + bk) * N1 + n0 + bnoff, 0, 1);
    }
    __syncthreads();
#pragma unroll
    for (int kk = 0; kk < 2; ++kk) {
      // hoist ALL fragments of this k-substep so the ds_load_b128 clause
      // overlaps with the WMMA pipe instead of stalling on s_wait_dscnt 0
      bf16x16 bfr[2], afr[4];
#pragma unroll
      for (int nt = 0; nt < 2; ++nt)
        bfr[nt] = ld_bfrag(&Bt[(wvN + nt * 16 + ln) * 64 + kk * 32 + hi * 8]);
#pragma unroll
      for (int mt = 0; mt < 4; ++mt)
        afr[mt] = ld_afrag(&At[(wvM + mt * 16 + ln) * 64 + kk * 32 + hi * 8]);
#pragma unroll
      for (int mt = 0; mt < 4; ++mt)
#pragma unroll
        for (int nt = 0; nt < 2; ++nt)
          acc[mt][nt] = wmma_bf16(afr[mt], bfr[nt], acc[mt][nt]);
    }
  }

#pragma unroll
  for (int mt = 0; mt < 4; ++mt)
#pragma unroll
    for (int nt = 0; nt < 2; ++nt) {
      const int gn = n0 + wvN + nt * 16 + ln;
      const float bv = bias[gn];
      const int mat = gn >> 10, cc = gn & 1023, hh = cc >> 6, dd = cc & 63;
      __bf16* dstbuf = (mat == 0) ? qb : (mat == 1) ? kb : vb;
#pragma unroll
      for (int r = 0; r < 8; ++r) {
        const int row = m0 + wvM + mt * 16 + r + 8 * hi;
        const int bb = row >> 11, tt = row & (T_ - 1);
        dstbuf[(((size_t)bb * H_ + hh) * T_ + tt) * D_ + dd] = f2bf(acc[mt][nt][r] + bv);
      }
    }
}

// ---------- kernel 2: causal flash attention, one (b,h) x 128 q-rows / block ----------

__global__ __launch_bounds__(256)
void attn_kernel(const __bf16* __restrict__ qb, const __bf16* __restrict__ kb,
                 const __bf16* __restrict__ vb, __bf16* __restrict__ yb) {
  __shared__ __bf16 Kt[32 * 64];       // [key][d] row-major
  __shared__ __bf16 Vt[64 * 32];       // [d][key] transposed
  __shared__ __bf16 Pb[8 * 16 * 32];   // per-wave C->A layout bounce
  const int tid = threadIdx.x, lane = tid & 31, wv = tid >> 5;
  const int ln = lane & 15, hi = lane >> 4;
  const int bh = blockIdx.y;           // b*H + h
  const int b = bh >> 4, h = bh & 15;
  const int qbase = blockIdx.x * 128;
  const int qr0 = qbase + wv * 16;
  const size_t headOff = (size_t)bh * T_ * D_;

  // Q fragments for this wave's 16 rows (k-dim = D = 64 -> 2 fragments)
  const __bf16* qp = qb + headOff + (size_t)(qr0 + ln) * D_;
  const bf16x16 qf0 = ld_afrag(qp + 0  + hi * 8);
  const bf16x16 qf1 = ld_afrag(qp + 32 + hi * 8);

  f32x8 O[4] = {};
  float mrow[8], lrow[8];
#pragma unroll
  for (int r = 0; r < 8; ++r) { mrow[r] = -1e30f; lrow[r] = 0.f; }

  const int krow = tid >> 3;           // 0..31 keys
  const int kdoff = (tid & 7) * 8;     // 0..56 d

  const int nkb = (qbase + 128) >> 5;
  for (int kblk = 0; kblk < nkb; ++kblk) {
    const int s0 = kblk << 5;
    __syncthreads();
    { // cooperative stage of K (row-major) and V (transposed)
      bf16x8 kv = *(const bf16x8*)(kb + headOff + (size_t)(s0 + krow) * D_ + kdoff);
      *(bf16x8*)(&Kt[krow * 64 + kdoff]) = kv;
      bf16x8 vv = *(const bf16x8*)(vb + headOff + (size_t)(s0 + krow) * D_ + kdoff);
#pragma unroll
      for (int i = 0; i < 8; ++i) Vt[(kdoff + i) * 32 + krow] = vv[i];
    }
    if (kblk + 1 < nkb) { // prefetch next key block
      __builtin_prefetch(kb + headOff + (size_t)(s0 + 32 + krow) * D_ + kdoff, 0, 1);
      __builtin_prefetch(vb + headOff + (size_t)(s0 + 32 + krow) * D_ + kdoff, 0, 1);
    }
    __syncthreads();
    if (s0 > qr0 + 15) continue;       // wave-uniform: fully masked block

    // S = Q K^T  (two 16x16 score tiles, two k-steps of 32 each)
    bf16x16 kf[4];
#pragma unroll
    for (int kk = 0; kk < 2; ++kk) {
      kf[kk * 2 + 0] = ld_bfrag(&Kt[ln * 64 + kk * 32 + hi * 8]);
      kf[kk * 2 + 1] = ld_bfrag(&Kt[(16 + ln) * 64 + kk * 32 + hi * 8]);
    }
    f32x8 S0 = {}, S1 = {};
    S0 = wmma_bf16(qf0, kf[0], S0);
    S1 = wmma_bf16(qf0, kf[1], S1);
    S0 = wmma_bf16(qf1, kf[2], S0);
    S1 = wmma_bf16(qf1, kf[3], S1);

    const bool needMask = (s0 + 31 > qr0);
    const int c0 = s0 + ln, c1 = c0 + 16;
    float P0[8], P1[8];
#pragma unroll
    for (int r = 0; r < 8; ++r) {
      float a0 = S0[r] * SCALE, a1 = S1[r] * SCALE;
      if (needMask) {
        const int qr = qr0 + r + 8 * hi;
        if (c0 > qr) a0 = -1e30f;
        if (c1 > qr) a1 = -1e30f;
      }
      // row max across the 16 lanes of this half-wave
      float mx = fmaxf(a0, a1);
      mx = fmaxf(mx, __shfl_xor(mx, 1, 32));
      mx = fmaxf(mx, __shfl_xor(mx, 2, 32));
      mx = fmaxf(mx, __shfl_xor(mx, 4, 32));
      mx = fmaxf(mx, __shfl_xor(mx, 8, 32));
      const float mnew  = fmaxf(mrow[r], mx);
      const float alpha = __expf(mrow[r] - mnew);
      const float p0 = __expf(a0 - mnew), p1 = __expf(a1 - mnew);
      float rs = p0 + p1;
      rs += __shfl_xor(rs, 1, 32);
      rs += __shfl_xor(rs, 2, 32);
      rs += __shfl_xor(rs, 4, 32);
      rs += __shfl_xor(rs, 8, 32);
      lrow[r] = lrow[r] * alpha + rs;
      mrow[r] = mnew;
      P0[r] = p0; P1[r] = p1;
#pragma unroll
      for (int nt = 0; nt < 4; ++nt) O[nt][r] *= alpha;
    }

    // C-layout -> A-layout via per-wave LDS (same-wave DS ops are in order)
    __bf16* pb = &Pb[wv * 512];
#pragma unroll
    for (int r = 0; r < 8; ++r) {
      const int m = r + 8 * hi;
      pb[m * 32 + ln]      = f2bf(P0[r]);
      pb[m * 32 + 16 + ln] = f2bf(P1[r]);
    }
    const bf16x16 pf = ld_afrag(&pb[ln * 32 + hi * 8]);
    bf16x16 vf[4];
#pragma unroll
    for (int nt = 0; nt < 4; ++nt)
      vf[nt] = ld_bfrag(&Vt[(nt * 16 + ln) * 32 + hi * 8]);
#pragma unroll
    for (int nt = 0; nt < 4; ++nt)
      O[nt] = wmma_bf16(pf, vf[nt], O[nt]);
  }

  // normalize and store y[b][t][h*64+d] (bf16)
#pragma unroll
  for (int r = 0; r < 8; ++r) lrow[r] = 1.f / lrow[r];
#pragma unroll
  for (int nt = 0; nt < 4; ++nt)
#pragma unroll
    for (int r = 0; r < 8; ++r) {
      const int t   = qr0 + r + 8 * hi;
      const int col = h * 64 + nt * 16 + ln;
      yb[((size_t)b * T_ + t) * C_ + col] = f2bf(O[nt][r] * lrow[r]);
    }
}

// ---------- kernel 3: out = y @ w_out + b_out (fp32 result) ----------

__global__ __launch_bounds__(256)
void out_gemm_kernel(const __bf16* __restrict__ y, const float* __restrict__ w,
                     const float* __restrict__ bias, float* __restrict__ out) {
  __shared__ __bf16 At[128 * 64];
  __shared__ __bf16 Bt[128 * 64];
  const int tid = threadIdx.x, lane = tid & 31, wv = tid >> 5;
  const int ln = lane & 15, hi = lane >> 4;
  const int n0 = blockIdx.x * 128, m0 = blockIdx.y * 128;
  const int wvM = (wv >> 2) * 64, wvN = (wv & 3) * 32;

  const int arow = tid >> 1, akoff = (tid & 1) * 32;
  const int bk = tid >> 2,  bnoff = (tid & 3) * 32;

  f32x8 acc[4][2] = {};

  for (int kb0 = 0; kb0 < C_; kb0 += 64) {
    __syncthreads();
    { // stage A (already bf16), 16B copies
      const bf16x8* src = (const bf16x8*)(y + (size_t)(m0 + arow) * C_ + kb0 + akoff);
      bf16x8* dst = (bf16x8*)(&At[arow * 64 + akoff]);
#pragma unroll
      for (int q = 0; q < 4; ++q) dst[q] = src[q];
    }
    { // stage B transposed (fp32 -> bf16)
      const float4* src = (const float4*)(w + (size_t)(kb0 + bk) * C_ + n0 + bnoff);
#pragma unroll
      for (int q = 0; q < 8; ++q) {
        float4 v = src[q];
        Bt[(bnoff + q * 4 + 0) * 64 + bk] = f2bf(v.x);
        Bt[(bnoff + q * 4 + 1) * 64 + bk] = f2bf(v.y);
        Bt[(bnoff + q * 4 + 2) * 64 + bk] = f2bf(v.z);
        Bt[(bnoff + q * 4 + 3) * 64 + bk] = f2bf(v.w);
      }
    }
    if (kb0 + 64 < C_) {
      __builtin_prefetch(y + (size_t)(m0 + arow) * C_ + kb0 + 64 + akoff, 0, 1);
      __builtin_prefetch(w + (size_t)(kb0 + 64 + bk) * C_ + n0 + bnoff, 0, 1);
    }
    __syncthreads();
#pragma unroll
    for (int kk = 0; kk < 2; ++kk) {
      bf16x16 bfr[2], afr[4];
#pragma unroll
      for (int nt = 0; nt < 2; ++nt)
        bfr[nt] = ld_bfrag(&Bt[(wvN + nt * 16 + ln) * 64 + kk * 32 + hi * 8]);
#pragma unroll
      for (int mt = 0; mt < 4; ++mt)
        afr[mt] = ld_afrag(&At[(wvM + mt * 16 + ln) * 64 + kk * 32 + hi * 8]);
#pragma unroll
      for (int mt = 0; mt < 4; ++mt)
#pragma unroll
        for (int nt = 0; nt < 2; ++nt)
          acc[mt][nt] = wmma_bf16(afr[mt], bfr[nt], acc[mt][nt]);
    }
  }

#pragma unroll
  for (int mt = 0; mt < 4; ++mt)
#pragma unroll
    for (int nt = 0; nt < 2; ++nt) {
      const int gn = n0 + wvN + nt * 16 + ln;
      const float bv = bias[gn];
#pragma unroll
      for (int r = 0; r < 8; ++r) {
        const int row = m0 + wvM + mt * 16 + r + 8 * hi;
        out[(size_t)row * C_ + gn] = acc[mt][nt][r] + bv;
      }
    }
}

// ---------- launch ----------

extern "C" void kernel_launch(void* const* d_in, const int* in_sizes, int n_in,
                              void* d_out, int out_size, void* d_ws, size_t ws_size,
                              hipStream_t stream) {
  (void)in_sizes; (void)n_in; (void)out_size; (void)ws_size;
  const float* x     = (const float*)d_in[0];
  const float* w_qkv = (const float*)d_in[1];
  const float* b_qkv = (const float*)d_in[2];
  const float* w_out = (const float*)d_in[3];
  const float* b_out = (const float*)d_in[4];
  float* out = (float*)d_out;

  const size_t HS = (size_t)B_ * H_ * T_ * D_;  // 8,388,608 elems
  __bf16* qb = (__bf16*)d_ws;
  __bf16* kb = qb + HS;
  __bf16* vb = kb + HS;
  __bf16* yb = vb + HS;                          // total 64 MB of ws

  qkv_gemm_kernel<<<dim3(N1 / 128, M_ / 128), 256, 0, stream>>>(x, w_qkv, b_qkv, qb, kb, vb);
  attn_kernel<<<dim3(T_ / 128, B_ * H_), 256, 0, stream>>>(qb, kb, vb, yb);
  out_gemm_kernel<<<dim3(C_ / 128, M_ / 128), 256, 0, stream>>>(yb, w_out, b_out, out);
}
